// decoderLSTM_15522011807916
// MI455X (gfx1250) — matrix operand order repaired and checked by reading, not compiled, not run on hardware
//
#include <hip/hip_runtime.h>
#include <math.h>

#define H  1024
#define NB 256
#define DD 256
#define TT 128

typedef __attribute__((ext_vector_type(16))) __bf16 v16bf;
typedef __attribute__((ext_vector_type(8)))  float  v8f;

// ---- branch-free transcendentals (native v_tanh_f32 when available) ----
__device__ __forceinline__ float fast_tanh(float x) {
#if __has_builtin(__builtin_amdgcn_tanh_f32)
    return __builtin_amdgcn_tanh_f32(x);
#elif __has_builtin(__builtin_amdgcn_tanhf)
    return __builtin_amdgcn_tanhf(x);
#else
    // clamped, branch-free: tanh(x) = (e^{2x}-1)/(e^{2x}+1)
    float xc = fminf(fmaxf(x, -15.0f), 15.0f);
    float e  = __expf(2.0f * xc);                 // v_exp_f32
    return (e - 1.0f) * __builtin_amdgcn_rcpf(e + 1.0f);
#endif
}
__device__ __forceinline__ float fast_sigmoid(float x) {
    return 0.5f + 0.5f * fast_tanh(0.5f * x);     // one TRANS op
}

// ---- WMMA fragment address helpers (CDNA5 ISA 7.12.2, wave32) ----
// A-matrix 16x32 bf16 fragment: lanes 0-15 hold M, K={0..7,16..23}; lanes 16-31 K={8..15,24..31}.
// Stored fragment-major: frag*512 + lane*16 + vgpr*2 + half   (bf16 element units; 32B/lane)
__device__ __forceinline__ int aFragElem(int m, int k, int ktiles) {
    int frag = (m >> 4) * ktiles + (k >> 5);
    int kk   = k & 31;
    int lane = (m & 15) + ((kk & 8) ? 16 : 0);
    int vg   = ((kk & 16) ? 4 : 0) + ((kk & 7) >> 1);
    int p    = kk & 1;
    return frag * 512 + lane * 16 + vg * 2 + p;
}
// B-matrix 32x16 bf16 fragment: lanes 0-15 hold N, K={0..7,16..23} (2/VGPR); lanes 16-31 K={8..15,24..31}.
__device__ __forceinline__ int bFragElem(int k, int n, int ktiles) {
    int frag = (n >> 4) * ktiles + (k >> 5);
    int kk   = k & 31;
    int lane = (n & 15) + ((kk & 8) ? 16 : 0);
    int vg   = ((kk & 16) ? 4 : 0) + ((kk & 7) >> 1);
    int p    = kk & 1;
    return frag * 512 + lane * 16 + vg * 2 + p;
}

// ---- Prologue: convert weights to bf16 fragment layout ----
// gates = h @ W^T  =>  B[k][n] = W[n][k].  WcF = (W_ih + W_hh) for steps >= 1.
__global__ void prep_weights(const float* __restrict__ Wih, const float* __restrict__ Whh,
                             __bf16* __restrict__ WihF, __bf16* __restrict__ WcF) {
    int idx = blockIdx.x * 256 + threadIdx.x;   // over 4H*H
    int n = idx >> 10, k = idx & 1023;
    float a = Wih[idx], b = Whh[idx];
    int off = bFragElem(k, n, H / 32);
    WihF[off] = (__bf16)a;
    WcF[off]  = (__bf16)(a + b);
}
__global__ void prep_fc(const float* __restrict__ Wfc, __bf16* __restrict__ WfcF) {
    int idx = blockIdx.x * 256 + threadIdx.x;   // over D*H
    int d = idx >> 10, k = idx & 1023;
    WfcF[bFragElem(k, d, H / 32)] = (__bf16)Wfc[idx];
}
__global__ void prep_bias_h(const float* __restrict__ bih, const float* __restrict__ bhh,
                            float* __restrict__ bias,
                            const float* __restrict__ hT, __bf16* __restrict__ hA0) {
    int idx = blockIdx.x * 256 + threadIdx.x;   // over N*H
    if (idx < 4 * H) bias[idx] = bih[idx] + bhh[idx];
    int b = idx >> 10, k = idx & 1023;
    hA0[aFragElem(b, k, H / 32)] = (__bf16)hT[idx];
}

// ---- Output projection: out[b][d] = h[b][:] @ Wfc^T + b_fc, per-wave 64x16 tile ----
__device__ __forceinline__ void do_proj(const __bf16* __restrict__ hA,
                                        const __bf16* __restrict__ WfcF,
                                        const float* __restrict__ bfc,
                                        float* __restrict__ outP,
                                        int pb, int wave, int lane) {
    const int KT = H / 32;
    int m0 = (pb & 3) * 64;
    int d0 = (pb >> 2) * 64 + wave * 16;
    v8f z = {};
    v8f acc[4] = {z, z, z, z};
    int ntBase = d0 >> 4;
    for (int kt = 0; kt < KT; ++kt) {
        v16bf bm = *(const v16bf*)(WfcF + (ntBase * KT + kt) * 512 + lane * 16);
#pragma unroll
        for (int mt = 0; mt < 4; ++mt) {
            v16bf a = *(const v16bf*)(hA + (((m0 >> 4) + mt) * KT + kt) * 512 + lane * 16);
            acc[mt] = __builtin_amdgcn_wmma_f32_16x16x32_bf16(
                false, a, false, bm, (short)0, acc[mt], false, false);
        }
    }
    int lcol = lane & 15, lhi = lane >> 4;
    int d = d0 + lcol;
    float bb = bfc[d];
#pragma unroll
    for (int mt = 0; mt < 4; ++mt)
#pragma unroll
        for (int r = 0; r < 8; ++r) {
            int b = m0 + mt * 16 + r + 8 * lhi;
            outP[(size_t)b * DD + d] = acc[mt][r] + bb;
        }
}

// ---- One LSTM step: blocks [0,64) gates+pointwise; blocks [64,80) project h_{t-1} ----
__global__ __launch_bounds__(128) void lstm_step(
    const __bf16* __restrict__ hA, __bf16* __restrict__ hAn,
    const __bf16* __restrict__ WF, const float* __restrict__ bias,
    float* __restrict__ cbuf, int firstStep,
    const __bf16* __restrict__ WfcF, const float* __restrict__ bfc,
    float* __restrict__ outPrev) {
    int wave = threadIdx.x >> 5;
    int lane = threadIdx.x & 31;
    int wg   = blockIdx.x;
    const int KT = H / 32;                       // 32 k-chunks

    if (wg >= 64) {                              // projection of previous h
        do_proj(hA, WfcF, bfc, outPrev, wg - 64, wave, lane);
        return;
    }

    int m0 = (wg & 3) * 64;                      // batch tile base
    int hc = (wg >> 2) * 64 + wave * 16;         // hidden-column base (this wave)

    v8f z = {};
    v8f acc[4][4];                               // [gate][m-tile], 16x16 f32 each
#pragma unroll
    for (int G = 0; G < 4; ++G)
#pragma unroll
        for (int mt = 0; mt < 4; ++mt) acc[G][mt] = z;

    for (int kt = 0; kt < KT; ++kt) {
        v16bf a[4];
#pragma unroll
        for (int mt = 0; mt < 4; ++mt)
            a[mt] = *(const v16bf*)(hA + (((m0 >> 4) + mt) * KT + kt) * 512 + lane * 16);
        v16bf bm[4];
#pragma unroll
        for (int G = 0; G < 4; ++G) {
            int nt = (G * H + hc) >> 4;          // same hidden cols across all 4 gates
            bm[G] = *(const v16bf*)(WF + (nt * KT + kt) * 512 + lane * 16);
        }
#pragma unroll
        for (int G = 0; G < 4; ++G)
#pragma unroll
            for (int mt = 0; mt < 4; ++mt)
                acc[G][mt] = __builtin_amdgcn_wmma_f32_16x16x32_bf16(
                    false, a[mt], false, bm[G], (short)0, acc[G][mt], false, false);
    }

    // Pointwise LSTM cell: i/f/g/o for identical (b,j) sit in matching (vgpr,lane) slots.
    int lcol = lane & 15, lhi = lane >> 4;
    int j = hc + lcol;
    float bi = bias[j], bf = bias[H + j], bg = bias[2 * H + j], bo = bias[3 * H + j];

    // Strength-reduced A-fragment store offsets for h (lane/sub parts are loop-invariant):
    // elem = ((m>>4)*KT + (j>>5))*512 + ((m&15) + laneAdd)*16 + sub*2 + p
    int kk      = j & 31;
    int laneAdd = (kk & 8) ? 16 : 0;
    int sub     = ((kk & 16) ? 4 : 0) + ((kk & 7) >> 1);
    int cbase   = (j >> 5) * 512 + sub * 2 + (kk & 1);
    int mrowHi  = 8 * lhi;

#pragma unroll
    for (int mt = 0; mt < 4; ++mt) {
        int fragRow = (((m0 >> 4) + mt) * KT) * 512;     // A frag base for this m-tile row
#pragma unroll
        for (int r = 0; r < 8; ++r) {
            int b = m0 + mt * 16 + r + mrowHi;
            float iv = fast_sigmoid(acc[0][mt][r] + bi);
            float fv = fast_sigmoid(acc[1][mt][r] + bf);
            float gv = fast_tanh(acc[2][mt][r] + bg);
            float ov = fast_sigmoid(acc[3][mt][r] + bo);
            float cold = firstStep ? 0.0f : cbuf[(size_t)b * H + j];
            float cn = fv * cold + iv * gv;
            cbuf[(size_t)b * H + j] = cn;
            float hv = ov * fast_tanh(cn);
            hAn[fragRow + cbase + (r + mrowHi + laneAdd) * 16] = (__bf16)hv;
        }
    }
}

__global__ __launch_bounds__(128) void proj_only(const __bf16* __restrict__ hA,
                                                 const __bf16* __restrict__ WfcF,
                                                 const float* __restrict__ bfc,
                                                 float* __restrict__ outP) {
    do_proj(hA, WfcF, bfc, outP, blockIdx.x, threadIdx.x >> 5, threadIdx.x & 31);
}

extern "C" void kernel_launch(void* const* d_in, const int* in_sizes, int n_in,
                              void* d_out, int out_size, void* d_ws, size_t ws_size,
                              hipStream_t stream) {
    (void)in_sizes; (void)n_in; (void)out_size; (void)ws_size;
    const float* hT  = (const float*)d_in[0];
    // d_in[1] = t (device scalar); fixed T = 128 at compile time.
    const float* Wih = (const float*)d_in[2];
    const float* Whh = (const float*)d_in[3];
    const float* bih = (const float*)d_in[4];
    const float* bhh = (const float*)d_in[5];
    const float* Wfc = (const float*)d_in[6];
    const float* bfc = (const float*)d_in[7];
    float* out = (float*)d_out;

    char* ws = (char*)d_ws;
    size_t off = 0;
    __bf16* WihF = (__bf16*)(ws + off); off += (size_t)4 * H * H * 2;  // 8 MB
    __bf16* WcF  = (__bf16*)(ws + off); off += (size_t)4 * H * H * 2;  // 8 MB
    __bf16* WfcF = (__bf16*)(ws + off); off += (size_t)DD * H * 2;     // 512 KB
    float*  bias = (float*)(ws + off);  off += (size_t)4 * H * 4;      // 16 KB
    __bf16* hA0  = (__bf16*)(ws + off); off += (size_t)NB * H * 2;     // 512 KB
    __bf16* hA1  = (__bf16*)(ws + off); off += (size_t)NB * H * 2;     // 512 KB
    float*  cbuf = (float*)(ws + off);  off += (size_t)NB * H * 4;     // 1 MB

    prep_weights<<<(4 * H * H) / 256, 256, 0, stream>>>(Wih, Whh, WihF, WcF);
    prep_fc<<<(DD * H) / 256, 256, 0, stream>>>(Wfc, WfcF);
    prep_bias_h<<<(NB * H) / 256, 256, 0, stream>>>(bih, bhh, bias, hT, hA0);

    __bf16* hbuf[2] = {hA0, hA1};
    for (int t = 0; t < TT; ++t) {
        const __bf16* hin  = hbuf[t & 1];        // h_{t-1}  (h_{-1} := hT)
        __bf16*       hout = hbuf[(t + 1) & 1];  // h_t
        const __bf16* W    = (t == 0) ? WihF : WcF;
        int doProj = (t > 0);
        float* outPrev = doProj ? (out + (size_t)(t - 1) * NB * DD) : nullptr;
        lstm_step<<<doProj ? 80 : 64, 128, 0, stream>>>(
            hin, hout, W, bias, cbuf, (t == 0) ? 1 : 0, WfcF, bfc, outPrev);
    }
    // h_{127} lives in hbuf[0]
    proj_only<<<16, 128, 0, stream>>>(hbuf[0], WfcF, bfc, out + (size_t)(TT - 1) * NB * DD);
}
